// selfAttentionBidirectional_42588895707870
// MI455X (gfx1250) — matrix operand order repaired and checked by reading, compiled, not run on hardware
//
#include <hip/hip_runtime.h>
#include <math.h>

typedef __attribute__((ext_vector_type(16))) _Float16 v16h;
typedef __attribute__((ext_vector_type(2)))  _Float16 v2h;
typedef __attribute__((ext_vector_type(8)))  float    v8f;

#if __has_builtin(__builtin_amdgcn_tensor_load_to_lds)
#define USE_TDM 1
#warning "CDNA5 probe: TDM path active (__builtin_amdgcn_tensor_load_to_lds found)"
typedef unsigned int tdm_v4u __attribute__((ext_vector_type(4)));
typedef int          tdm_v8i __attribute__((ext_vector_type(8)));
typedef int          tdm_v4i __attribute__((ext_vector_type(4)));
#else
#define USE_TDM 0
#warning "CDNA5 probe: TDM builtin NOT found; direct global-load fallback"
#endif

#define B_     4
#define S_     4096
#define DIN_   512
#define DOUT_  64
#define CHUNK_ 64
#define NWAVE_ 4      // q-tiles (waves) per workgroup sharing one K/V staging

// ---------------------------------------------------------------------------
// Phase 1: QKV projection.  One wave per 16-row tile per matrix.
// q, k stored f16 row-major [B,S,64]; v stored TRANSPOSED f16 [B,64,S].
// ---------------------------------------------------------------------------
__global__ __launch_bounds__(32) void qkv_proj_wmma(
    const float* __restrict__ x,
    const float* __restrict__ Wq,
    const float* __restrict__ Wk,
    const float* __restrict__ Wv,
    _Float16* __restrict__ qh,
    _Float16* __restrict__ kh,
    _Float16* __restrict__ vT)
{
    const int lane = threadIdx.x & 31;
    const int m    = lane & 15;
    const int g    = lane >> 4;
    const int tile = blockIdx.x;
    const int b    = tile / (S_ / 16);
    const int row0 = (tile % (S_ / 16)) * 16;
    const int which = blockIdx.y;        // 0=Q 1=K 2=V

    const float* W = (which == 0) ? Wq : (which == 1) ? Wk : Wv;

    v8f acc[4] = {};
    const float* xrow = x + ((size_t)(b * S_ + row0 + m)) * DIN_;

    for (int kc = 0; kc < DIN_ / 32; ++kc) {
        v16h a;                                    // A: x[16x32] f32->f16
#pragma unroll
        for (int v = 0; v < 8; ++v) {
            const int k = kc * 32 + (v >> 2) * 16 + g * 8 + (v & 3) * 2;
            const float2 f = *(const float2*)(xrow + k);
            a[2 * v]     = (_Float16)f.x;
            a[2 * v + 1] = (_Float16)f.y;
        }
#pragma unroll
        for (int nt = 0; nt < 4; ++nt) {
            const float* wrow = W + (size_t)(nt * 16 + m) * DIN_;
            v16h bf;                               // B(k,n) = W[n,k]
#pragma unroll
            for (int j = 0; j < 8; ++j) {
                const int k = kc * 32 + (j >> 2) * 16 + g * 8 + (j & 3) * 2;
                const float2 f = *(const float2*)(wrow + k);
                bf[2 * j]     = (_Float16)f.x;
                bf[2 * j + 1] = (_Float16)f.y;
            }
            acc[nt] = __builtin_amdgcn_wmma_f32_16x16x32_f16(
                false, a, false, bf, (short)0, acc[nt], false, false);
        }
    }

#pragma unroll
    for (int nt = 0; nt < 4; ++nt)
#pragma unroll
        for (int r = 0; r < 8; ++r) {
            const int row = row0 + r + 8 * g;
            const int n   = nt * 16 + m;
            const _Float16 val = (_Float16)acc[nt][r];
            if (which == 2)
                vT[((size_t)(b * DOUT_ + n)) * S_ + row] = val;
            else if (which == 1)
                kh[((size_t)(b * S_ + row)) * DOUT_ + n] = val;
            else
                qh[((size_t)(b * S_ + row)) * DOUT_ + n] = val;
        }
}

#if USE_TDM
// ---------------------------------------------------------------------------
// TDM: DMA one 64x64 f16 tile (row-major, row stride = stride0 elements)
// from global into LDS.  D# per CDNA5 ISA Ch.8.
// ---------------------------------------------------------------------------
__device__ __forceinline__ void tdm_load_tile_64x64_f16(
    const _Float16* gsrc, const _Float16* lds_dst, unsigned long long stride0)
{
    const unsigned long long ga = (unsigned long long)(size_t)gsrc;
    const unsigned lds = (unsigned)(size_t)lds_dst;  // flat low 32 = LDS offset

    tdm_v4u g0;
    g0[0] = 1u;                                      // count=1, user descriptor
    g0[1] = lds;                                     // lds_addr
    g0[2] = (unsigned)(ga & 0xFFFFFFFFull);          // global_addr[31:0]
    g0[3] = (unsigned)((ga >> 32) & 0x01FFFFFFull)   // global_addr[56:32]
            | (2u << 30);                            // type = 2 ("image")

    tdm_v8i g1;
    g1[0] = (int)(1u << 16);                         // data_size = 1 (2 bytes)
    g1[1] = (int)(64u << 16);                        // tensor_dim0[15:0] = 64
    g1[2] = (int)(64u << 16);                        // tensor_dim1[15:0] = 64
    g1[3] = (int)(64u << 16);                        // tile_dim0 = 64
    g1[4] = 64;                                      // tile_dim1 = 64, tile_dim2=0
    g1[5] = (int)(unsigned)(stride0 & 0xFFFFFFFFull);        // dim0 stride lo
    g1[6] = (int)(unsigned)((stride0 >> 32) & 0xFFFFull);    // dim0 stride hi
    g1[7] = 0;                                       // dim1 stride (unused, 2D)

    tdm_v4i gz = {0, 0, 0, 0};
#if __clang_major__ >= 23
    tdm_v8i gz8 = {0, 0, 0, 0, 0, 0, 0, 0};
    __builtin_amdgcn_tensor_load_to_lds(g0, g1, gz, gz, gz8, 0);
#else
    __builtin_amdgcn_tensor_load_to_lds(g0, g1, gz, gz, 0);
#endif
}
#endif

// ---------------------------------------------------------------------------
// Phase 2: flash attention.  4 waves per workgroup, each owning a 16-row
// query tile; one shared TDM double-buffered K/V staging per workgroup.
// ---------------------------------------------------------------------------
__global__ __launch_bounds__(32 * NWAVE_) void flash_attn_wmma(
    const _Float16* __restrict__ qh,
    const _Float16* __restrict__ kh,
    const _Float16* __restrict__ vT,
    float* __restrict__ out)
{
    __shared__ _Float16 pshm[NWAVE_][16][CHUNK_];    // per-wave P relay (8 KB)
#if USE_TDM
    __shared__ _Float16 kbuf[2][CHUNK_][DOUT_];      // [key][d]   (16 KB)
    __shared__ _Float16 vbuf[2][DOUT_][CHUNK_];      // [d][key]   (16 KB)
#endif

    const int lane = threadIdx.x & 31;
    const int w    = threadIdx.x >> 5;               // wave id in workgroup
    const int m    = lane & 15;
    const int g    = lane >> 4;
    const int tile = blockIdx.x;                     // 0 .. B*S/64-1
    const int b    = tile / (S_ / 64);
    const int row0 = (tile % (S_ / 64)) * 64 + w * 16;

    // Q fragments (A layout), resident all kernel
    v16h aq[2];
    {
        const _Float16* qrow = qh + ((size_t)(b * S_ + row0 + m)) * DOUT_;
#pragma unroll
        for (int kc = 0; kc < 2; ++kc)
#pragma unroll
            for (int v = 0; v < 8; ++v) {
                const int k = kc * 32 + (v >> 2) * 16 + g * 8 + (v & 3) * 2;
                const v2h h = *(const v2h*)(qrow + k);
                aq[kc][2 * v]     = h.x;
                aq[kc][2 * v + 1] = h.y;
            }
    }

    v8f o[4] = {};
    float mrow[8], lsum[8];
#pragma unroll
    for (int r = 0; r < 8; ++r) { mrow[r] = -3.0e38f; lsum[r] = 0.0f; }

    const float SC = 0.125f * 1.44269504088896340736f;   // scale * log2(e)

#if USE_TDM
    if (w == 0) {                                    // wave 0 drives the TDM
        tdm_load_tile_64x64_f16(kh + (size_t)(b * S_) * DOUT_,
                                &kbuf[0][0][0], DOUT_);
        tdm_load_tile_64x64_f16(vT + (size_t)(b * DOUT_) * S_,
                                &vbuf[0][0][0], S_);
    }
#endif

    const int NCH = S_ / CHUNK_;
    for (int jc = 0; jc < NCH; ++jc) {
        const int key0 = jc * CHUNK_;
        const int cur  = jc & 1;

#if USE_TDM
        if (w == 0) {
            if (jc + 1 < NCH) {
                // wave0's own LDS reads of buf[cur^1] must have completed
                // (other waves' reads finished at last iteration's barrier)
                asm volatile("s_wait_dscnt 0x0" ::: "memory");
                tdm_load_tile_64x64_f16(
                    kh + (size_t)(b * S_ + key0 + CHUNK_) * DOUT_,
                    &kbuf[cur ^ 1][0][0], DOUT_);
                tdm_load_tile_64x64_f16(
                    vT + (size_t)(b * DOUT_) * S_ + key0 + CHUNK_,
                    &vbuf[cur ^ 1][0][0], S_);
                __builtin_amdgcn_s_wait_tensorcnt(2);  // chunk jc resident
            } else {
                __builtin_amdgcn_s_wait_tensorcnt(0);
            }
        }
#endif
        __syncthreads();                             // residency visible to all

        // ---- scores S = Q.K^T for 64 keys: 4 N-tiles x 2 K-steps ----------
        v8f s[4] = {};
#pragma unroll
        for (int nt = 0; nt < 4; ++nt) {
#if USE_TDM
            const _Float16* krow = &kbuf[cur][nt * 16 + m][0];
#else
            const _Float16* krow =
                kh + ((size_t)(b * S_ + key0 + nt * 16 + m)) * DOUT_;
#endif
#pragma unroll
            for (int kc = 0; kc < 2; ++kc) {
                v16h bk;
#pragma unroll
                for (int j = 0; j < 8; ++j) {
                    const int d = kc * 32 + (j >> 2) * 16 + g * 8 + (j & 3) * 2;
                    const v2h h = *(const v2h*)(krow + d);
                    bk[2 * j]     = h.x;
                    bk[2 * j + 1] = h.y;
                }
                s[nt] = __builtin_amdgcn_wmma_f32_16x16x32_f16(
                    false, aq[kc], false, bk, (short)0, s[nt], false, false);
            }
        }

        // ---- online softmax over 16x64 (row M = r+8g lives in one half-wave)
        float rmax[8];
#pragma unroll
        for (int r = 0; r < 8; ++r)
            rmax[r] = fmaxf(fmaxf(s[0][r], s[1][r]), fmaxf(s[2][r], s[3][r]));
#pragma unroll
        for (int off = 1; off < 16; off <<= 1)
#pragma unroll
            for (int r = 0; r < 8; ++r)
                rmax[r] = fmaxf(rmax[r], __shfl_xor(rmax[r], off, 32));

        float corr[8], rsum[8];
#pragma unroll
        for (int r = 0; r < 8; ++r) {
            const float mn = fmaxf(mrow[r], rmax[r]);
            corr[r] = exp2f((mrow[r] - mn) * SC);
            mrow[r] = mn;
            float acc = 0.0f;
#pragma unroll
            for (int t = 0; t < 4; ++t) {
                const float p = exp2f((s[t][r] - mn) * SC);
                acc += p;
                pshm[w][r + 8 * g][t * 16 + m] = (_Float16)p;
            }
            rsum[r] = acc;
        }
#pragma unroll
        for (int off = 1; off < 16; off <<= 1)
#pragma unroll
            for (int r = 0; r < 8; ++r)
                rsum[r] += __shfl_xor(rsum[r], off, 32);
#pragma unroll
        for (int r = 0; r < 8; ++r) {
            lsum[r] = lsum[r] * corr[r] + rsum[r];
#pragma unroll
            for (int nt = 0; nt < 4; ++nt)
                o[nt][r] *= corr[r];
        }

        // per-wave P relay: DS ops from one wave stay in order (RAW safe),
        // just keep the compiler from reordering around it
        asm volatile("" ::: "memory");

        // ---- O += P.V : P as two A-fragments (16x64) ----------------------
        v16h ap[2];
#pragma unroll
        for (int kc = 0; kc < 2; ++kc)
#pragma unroll
            for (int v = 0; v < 8; ++v) {
                const int k = kc * 32 + (v >> 2) * 16 + g * 8 + (v & 3) * 2;
                const v2h h = *(const v2h*)(&pshm[w][m][k]);
                ap[kc][2 * v]     = h.x;
                ap[kc][2 * v + 1] = h.y;
            }

#pragma unroll
        for (int nt = 0; nt < 4; ++nt) {
#if USE_TDM
            const _Float16* vrow = &vbuf[cur][nt * 16 + m][0];
#else
            const _Float16* vrow =
                vT + ((size_t)(b * DOUT_ + nt * 16 + m)) * S_ + key0;
#endif
#pragma unroll
            for (int kc = 0; kc < 2; ++kc) {
                v16h bv;
#pragma unroll
                for (int j = 0; j < 8; ++j) {
                    const int kk = kc * 32 + (j >> 2) * 16 + g * 8 + (j & 3) * 2;
                    const v2h h = *(const v2h*)(vrow + kk);
                    bv[2 * j]     = h.x;
                    bv[2 * j + 1] = h.y;
                }
                o[nt] = __builtin_amdgcn_wmma_f32_16x16x32_f16(
                    false, ap[kc], false, bv, (short)0, o[nt], false, false);
            }
        }

        __syncthreads();          // all waves done with buf[cur] before it is
                                  // re-targeted by wave0's TDM next iteration
    }

    // ---- normalize and store ----------------------------------------------
#pragma unroll
    for (int r = 0; r < 8; ++r) {
        const float inv = 1.0f / lsum[r];
        const int row = row0 + r + 8 * g;
#pragma unroll
        for (int nt = 0; nt < 4; ++nt)
            out[((size_t)(b * S_ + row)) * DOUT_ + nt * 16 + m] =
                o[nt][r] * inv;
    }
}

// ---------------------------------------------------------------------------
extern "C" void kernel_launch(void* const* d_in, const int* in_sizes, int n_in,
                              void* d_out, int out_size, void* d_ws, size_t ws_size,
                              hipStream_t stream) {
    (void)in_sizes; (void)n_in; (void)out_size; (void)ws_size;

    const float* x  = (const float*)d_in[0];
    const float* Wq = (const float*)d_in[1];
    const float* Wk = (const float*)d_in[2];
    const float* Wv = (const float*)d_in[3];
    float* out = (float*)d_out;

    _Float16* qh = (_Float16*)d_ws;
    _Float16* kh = qh + (size_t)B_ * S_ * DOUT_;
    _Float16* vT = kh + (size_t)B_ * S_ * DOUT_;

    dim3 gridP(B_ * S_ / 16, 3);
    qkv_proj_wmma<<<gridP, 32, 0, stream>>>(x, Wq, Wk, Wv, qh, kh, vT);
    flash_attn_wmma<<<B_ * S_ / 64, 32 * NWAVE_, 0, stream>>>(qh, kh, vT, out);
}
